// Transformer_15805479649981
// MI455X (gfx1250) — matrix operand order repaired and checked by reading, not compiled, not run on hardware
//
#include <hip/hip_runtime.h>
#include <hip/hip_bf16.h>
#include <math.h>

// Problem constants (match reference)
#define NBLK 4
#define HH   8
#define DD   1024
#define DKK  128
#define DFFN 4096
#define VV   32000
#define BB   2
#define LL   1024
#define MM   (BB*LL)   // 2048 tokens

typedef __attribute__((ext_vector_type(16))) __bf16 v16bf;
typedef __attribute__((ext_vector_type(8)))  __bf16 v8bf;
typedef __attribute__((ext_vector_type(8)))  float  v8f;

__device__ __forceinline__ unsigned short f2bf(float f) {
  unsigned int u = __float_as_uint(f);
  u += 0x7FFFu + ((u >> 16) & 1u);          // round-to-nearest-even
  return (unsigned short)(u >> 16);
}

// ---------------------------------------------------------------- cast fp32 -> bf16
__global__ void k_cast_bf16(const float* __restrict__ src,
                            unsigned short* __restrict__ dst, long long n) {
  long long i = (long long)blockIdx.x * blockDim.x + threadIdx.x;
  long long stride = (long long)gridDim.x * blockDim.x;
  for (; i < n; i += stride) dst[i] = f2bf(src[i]);
}

// ---------------------------------------------------------------- embedding gather
__global__ void k_embed(const float* __restrict__ emb,
                        const long long* __restrict__ ids,
                        float* __restrict__ xf, unsigned short* __restrict__ xb) {
  int t  = blockIdx.x;
  int id = (int)ids[t];
  for (int d = threadIdx.x; d < DD; d += blockDim.x) {
    float v = emb[(long long)d * VV + id];
    xf[(long long)t * DD + d] = v;
    xb[(long long)t * DD + d] = f2bf(v);
  }
}

// ---------------------------------------------------------------- batched bf16 WMMA GEMM
// C[z] = scale * A[z] x B[z](^T) + bias[z]  (optional relu; fp32 and/or bf16 out)
// Block tile 128x128, 8 waves, wave tile 32x64 = 2x4 v_wmma_f32_16x16x32_bf16.
#define TM 128
#define TN 128
#define TK 32
#define LSTR 40   // LDS row stride in ushorts: 80B = 20 banks -> conflict-free, 16B aligned

__global__ __launch_bounds__(256)
void k_gemm_bf16(const unsigned short* __restrict__ A,
                 const unsigned short* __restrict__ Bm,
                 const float* __restrict__ bias,
                 float* __restrict__ Cf,
                 unsigned short* __restrict__ Cb,
                 int M, int N, int K,
                 int lda, int ldb, int ldc,
                 long long aBatch, long long bBatch, long long cBatch,
                 long long biasBatch,
                 float scale, int relu, int transB)
{
  __shared__ __align__(16) unsigned short lA[TM * LSTR];   // A tile, row-major [m][k]
  __shared__ __align__(16) unsigned short lBt[TN * LSTR];  // B tile, TRANSPOSED [n][k]

  const int tid  = threadIdx.x;
  const int lane = tid & 31;          // wave32
  const int wid  = tid >> 5;          // 8 waves
  const int wr   = wid >> 1;          // 0..3 -> 32-row strip
  const int wc   = wid & 1;           // 0..1 -> 64-col strip

  const int row0 = blockIdx.y * TM;
  const int col0 = blockIdx.x * TN;
  const long long z = blockIdx.z;

  const unsigned short* Az = A  + z * aBatch;
  const unsigned short* Bz = Bm + z * bBatch;

  // --- per-thread tile-fill pointers (hoisted; advanced by TK per step) ---
  // A: 128 rows x 4 x (8 ushort) chunks = 512 b128 loads -> 2 per thread
  const int rA = tid >> 2;            // 0..63
  const int cA = (tid & 3) * 8;       // ushort column
  const unsigned short* gA0 = Az + (long long)(row0 + rA) * lda + cA;
  const unsigned short* gA1 = gA0 + (long long)64 * lda;
  unsigned short* sA0 = &lA[rA * LSTR + cA];
  unsigned short* sA1 = &lA[(rA + 64) * LSTR + cA];

  const unsigned short* gB0;
  const unsigned short* gB1;
  unsigned short* sB0 = nullptr;
  unsigned short* sB1 = nullptr;
  int kB = 0, cnB = 0;
  long long bStep;
  if (transB) {
    // B source is [N][K]: rows are n, contiguous in k -> direct b128 into lBt[n][k]
    const int nB = tid >> 2;          // 0..63
    const int cB = (tid & 3) * 8;
    gB0 = Bz + (long long)(col0 + nB) * ldb + cB;
    gB1 = gB0 + (long long)64 * ldb;
    sB0 = &lBt[nB * LSTR + cB];
    sB1 = &lBt[(nB + 64) * LSTR + cB];
    bStep = TK;
  } else {
    // B source is [K][N]: rows are k, contiguous in n -> b128 load, transpose on store
    kB  = tid >> 3;                   // 0..31
    cnB = (tid & 7) * 8;              // 0..56
    gB0 = Bz + (long long)kB * ldb + col0 + cnB;
    gB1 = gB0 + 64;
    bStep = (long long)TK * ldb;
  }

  v8f acc[2][4] = {};

  // ISA 7.12.2 lane mapping (wave32)
  const int m16 = lane & 15;
  const int kbA = (lane >> 4) << 3;   // A: half-wave K base 0/8 (runs at +0 and +16)
  const int kbB = (lane >> 4) << 3;   // B: half-wave K base 0/8 (runs at +0 and +16)

  for (int k0 = 0; k0 < K; k0 += TK) {
    const uint4 a0 = *(const uint4*)gA0;
    const uint4 a1 = *(const uint4*)gA1;
    const uint4 b0 = *(const uint4*)gB0;
    const uint4 b1 = *(const uint4*)gB1;
    *(uint4*)sA0 = a0;
    *(uint4*)sA1 = a1;
    if (transB) {
      *(uint4*)sB0 = b0;
      *(uint4*)sB1 = b1;
    } else {
      const unsigned short* p0 = (const unsigned short*)&b0;
      const unsigned short* p1 = (const unsigned short*)&b1;
      #pragma unroll
      for (int j = 0; j < 8; ++j) {
        lBt[(cnB + j) * LSTR + kB]        = p0[j];
        lBt[(cnB + 64 + j) * LSTR + kB]   = p1[j];
      }
    }
    gA0 += TK; gA1 += TK; gB0 += bStep; gB1 += bStep;
    if (k0 + TK < K) {                  // CDNA5 global_prefetch_b8 of next tiles
      __builtin_prefetch(gA0, 0, 1);
      __builtin_prefetch(gB0, 0, 1);
    }
    __syncthreads();

    // fragments: two aligned 16B LDS loads each, concatenated (no repacking VALU)
    v16bf av[2];
    #pragma unroll
    for (int i = 0; i < 2; ++i) {
      const unsigned short* p = &lA[(wr * 32 + i * 16 + m16) * LSTR + kbA];
      const v8bf lo = *(const v8bf*)p;
      const v8bf hi = *(const v8bf*)(p + 16);
      av[i] = __builtin_shufflevector(lo, hi, 0,1,2,3,4,5,6,7,8,9,10,11,12,13,14,15);
    }
    v16bf bv[4];
    #pragma unroll
    for (int j = 0; j < 4; ++j) {
      const unsigned short* p = &lBt[(wc * 64 + j * 16 + m16) * LSTR + kbB];
      const v8bf lo = *(const v8bf*)p;
      const v8bf hi = *(const v8bf*)(p + 16);
      bv[j] = __builtin_shufflevector(lo, hi, 0,1,2,3,4,5,6,7,8,9,10,11,12,13,14,15);
    }

    #pragma unroll
    for (int i = 0; i < 2; ++i)
      #pragma unroll
      for (int j = 0; j < 4; ++j)
        acc[i][j] = __builtin_amdgcn_wmma_f32_16x16x32_bf16(
            false, av[i], false, bv[j], (short)0, acc[i][j], false, false);

    __syncthreads();
  }

  // epilogue: D layout = VGPR r <-> row r + 8*(lane>=16), col = lane&15
  const int rhalf = (lane >> 4) << 3;
  #pragma unroll
  for (int i = 0; i < 2; ++i) {
    #pragma unroll
    for (int j = 0; j < 4; ++j) {
      const int gn = col0 + wc * 64 + j * 16 + m16;
      #pragma unroll
      for (int r = 0; r < 8; ++r) {
        const int gm = row0 + wr * 32 + i * 16 + rhalf + r;
        float val = acc[i][j][r] * scale;
        if (bias) val += bias[z * biasBatch + gn];
        if (relu) val = fmaxf(val, 0.f);
        const long long cidx = z * cBatch + (long long)gm * ldc + gn;
        if (Cf) Cf[cidx] = val;
        if (Cb) Cb[cidx] = f2bf(val);
      }
    }
  }
}

// ---------------------------------------------------------------- residual + LayerNorm
__global__ __launch_bounds__(256)
void k_add_ln(const float* __restrict__ yin, const float* __restrict__ res,
              const float* __restrict__ g, const float* __restrict__ b,
              float* __restrict__ xf, unsigned short* __restrict__ xb)
{
  __shared__ float red[256];
  const int t = blockIdx.x;
  const float* y = yin + (long long)t * DD;
  const float* r = res + (long long)t * DD;

  float s = 0.f;
  for (int d = threadIdx.x; d < DD; d += 256) s += y[d] + r[d];
  red[threadIdx.x] = s; __syncthreads();
  for (int o = 128; o > 0; o >>= 1) {
    if (threadIdx.x < o) red[threadIdx.x] += red[threadIdx.x + o];
    __syncthreads();
  }
  const float mean = red[0] * (1.0f / DD);
  __syncthreads();

  float s2 = 0.f;
  for (int d = threadIdx.x; d < DD; d += 256) {
    float v = y[d] + r[d] - mean; s2 += v * v;
  }
  red[threadIdx.x] = s2; __syncthreads();
  for (int o = 128; o > 0; o >>= 1) {
    if (threadIdx.x < o) red[threadIdx.x] += red[threadIdx.x + o];
    __syncthreads();
  }
  const float inv = rsqrtf(red[0] * (1.0f / DD) + 1e-5f);

  for (int d = threadIdx.x; d < DD; d += 256) {
    float v = (y[d] + r[d] - mean) * inv * g[d] + b[d];
    xf[(long long)t * DD + d] = v;
    xb[(long long)t * DD + d] = f2bf(v);
  }
}

// ---------------------------------------------------------------- softmax then post-mask
__global__ __launch_bounds__(256)
void k_softmax_mask(const float* __restrict__ S, unsigned short* __restrict__ att,
                    const int* __restrict__ pm)
{
  __shared__ float red[256];
  const int q = blockIdx.x;
  const int z = blockIdx.y;        // h*B + b
  const int b = z % BB;
  const float* row = S + ((long long)z * LL + q) * LL;
  unsigned short* orow = att + ((long long)z * LL + q) * LL;

  float mx = -3.4e38f;
  for (int s = threadIdx.x; s < LL; s += 256) mx = fmaxf(mx, row[s]);
  red[threadIdx.x] = mx; __syncthreads();
  for (int o = 128; o > 0; o >>= 1) {
    if (threadIdx.x < o) red[threadIdx.x] = fmaxf(red[threadIdx.x], red[threadIdx.x + o]);
    __syncthreads();
  }
  mx = red[0]; __syncthreads();

  float sum = 0.f;
  for (int s = threadIdx.x; s < LL; s += 256) sum += __expf(row[s] - mx);
  red[threadIdx.x] = sum; __syncthreads();
  for (int o = 128; o > 0; o >>= 1) {
    if (threadIdx.x < o) red[threadIdx.x] += red[threadIdx.x + o];
    __syncthreads();
  }
  const float inv = 1.f / red[0];

  const float pq = (float)pm[b * LL + q];
  for (int s = threadIdx.x; s < LL; s += 256) {
    float v = __expf(row[s] - mx) * inv;
    float m = (s <= q) ? pq * (float)pm[b * LL + s] : 0.f;
    orow[s] = f2bf(v * m);
  }
}

// ---------------------------------------------------------------- concat heads
__global__ void k_concat_heads(const unsigned short* __restrict__ O,
                               unsigned short* __restrict__ zc)
{
  long long i = (long long)blockIdx.x * blockDim.x + threadIdx.x;
  const long long tot = (long long)HH * BB * LL * DKK;
  if (i >= tot) return;
  const int n = (int)(i & (DKK - 1));
  const long long r = i >> 7;              // (h*B+b)*L + q
  const int q = (int)(r & (LL - 1));
  const long long hb = r >> 10;
  const int b = (int)(hb % BB);
  const int h = (int)(hb / BB);
  zc[(((long long)b * LL + q) * (HH * DKK)) + h * DKK + n] = O[i];
}

// ---------------------------------------------------------------- launcher
extern "C" void kernel_launch(void* const* d_in, const int* in_sizes, int n_in,
                              void* d_out, int out_size, void* d_ws, size_t ws_size,
                              hipStream_t stream)
{
  (void)in_sizes; (void)n_in; (void)out_size; (void)ws_size;

  const float*     emb  = (const float*)d_in[0];
  const float*     Wq   = (const float*)d_in[1];
  const float*     bq   = (const float*)d_in[2];
  const float*     Wk   = (const float*)d_in[3];
  const float*     bk   = (const float*)d_in[4];
  const float*     Wv   = (const float*)d_in[5];
  const float*     bv   = (const float*)d_in[6];
  const float*     Wo   = (const float*)d_in[7];
  const float*     bo   = (const float*)d_in[8];
  const float*     ln1g = (const float*)d_in[9];
  const float*     ln1b = (const float*)d_in[10];
  const float*     ln3g = (const float*)d_in[11];
  const float*     ln3b = (const float*)d_in[12];
  const float*     W1   = (const float*)d_in[13];
  const float*     b1   = (const float*)d_in[14];
  const float*     W2   = (const float*)d_in[15];
  const float*     b2   = (const float*)d_in[16];
  const float*     Wout = (const float*)d_in[17];
  const float*     bout = (const float*)d_in[18];
  const long long* ids  = (const long long*)d_in[19];   // int64 token ids
  const int*       pm   = (const int*)d_in[20];         // int32 pad mask
  float*           out  = (float*)d_out;

  char* w = (char*)d_ws;
  size_t off = 0;
  auto alloc = [&](size_t bytes) -> void* {
    void* p = (void*)(w + off);
    off += (bytes + 255) & ~(size_t)255;
    return p;
  };

  float*          xf    = (float*)         alloc((size_t)MM * DD * 4);
  unsigned short* xb    = (unsigned short*)alloc((size_t)MM * DD * 2);
  unsigned short* qb    = (unsigned short*)alloc((size_t)HH * MM * DKK * 2);
  unsigned short* kb    = (unsigned short*)alloc((size_t)HH * MM * DKK * 2);
  unsigned short* vb    = (unsigned short*)alloc((size_t)HH * MM * DKK * 2);
  float*          Smat  = (float*)         alloc((size_t)HH * BB * LL * LL * 4);
  unsigned short* attb  = (unsigned short*)alloc((size_t)HH * BB * LL * LL * 2);
  unsigned short* Ob    = (unsigned short*)alloc((size_t)HH * BB * LL * DKK * 2);
  unsigned short* zc    = (unsigned short*)alloc((size_t)MM * HH * DKK * 2);
  float*          tf    = (float*)         alloc((size_t)MM * DD * 4);
  unsigned short* h1b   = (unsigned short*)alloc((size_t)MM * DFFN * 2);
  unsigned short* wQb   = (unsigned short*)alloc((size_t)HH * DD * DKK * 2);
  unsigned short* wKb   = (unsigned short*)alloc((size_t)HH * DD * DKK * 2);
  unsigned short* wVb   = (unsigned short*)alloc((size_t)HH * DD * DKK * 2);
  unsigned short* wOb   = (unsigned short*)alloc((size_t)DD * DD * 2);
  unsigned short* w1b   = (unsigned short*)alloc((size_t)DD * DFFN * 2);
  unsigned short* w2b   = (unsigned short*)alloc((size_t)DFFN * DD * 2);
  unsigned short* woutb = (unsigned short*)alloc((size_t)DD * VV * 2);

  const float iscale = 0.08838834764831845f;  // 1/sqrt(DK=128)

  k_embed<<<MM, 256, 0, stream>>>(emb, ids, xf, xb);
  k_cast_bf16<<<4096, 256, 0, stream>>>(Wout, woutb, (long long)DD * VV);

  for (int i = 0; i < NBLK; ++i) {
    const long long wqkv = (long long)HH * DD * DKK;
    k_cast_bf16<<<2048, 256, 0, stream>>>(Wq + (long long)i * wqkv, wQb, wqkv);
    k_cast_bf16<<<2048, 256, 0, stream>>>(Wk + (long long)i * wqkv, wKb, wqkv);
    k_cast_bf16<<<2048, 256, 0, stream>>>(Wv + (long long)i * wqkv, wVb, wqkv);
    k_cast_bf16<<<2048, 256, 0, stream>>>(Wo + (long long)i * DD * DD, wOb, (long long)DD * DD);
    k_cast_bf16<<<2048, 256, 0, stream>>>(W1 + (long long)i * DD * DFFN, w1b, (long long)DD * DFFN);
    k_cast_bf16<<<2048, 256, 0, stream>>>(W2 + (long long)i * DFFN * DD, w2b, (long long)DFFN * DD);

    // QKV projections: batch over heads; Q/K/V stored as [H][B*L][DK] (bf16)
    dim3 gq(DKK / TN, MM / TM, HH);
    k_gemm_bf16<<<gq, 256, 0, stream>>>(xb, wQb, bq + (long long)i * HH * DKK,
        nullptr, qb, MM, DKK, DD, DD, DKK, DKK,
        0LL, (long long)DD * DKK, (long long)MM * DKK, (long long)DKK, 1.f, 0, 0);
    k_gemm_bf16<<<gq, 256, 0, stream>>>(xb, wKb, bk + (long long)i * HH * DKK,
        nullptr, kb, MM, DKK, DD, DD, DKK, DKK,
        0LL, (long long)DD * DKK, (long long)MM * DKK, (long long)DKK, 1.f, 0, 0);
    k_gemm_bf16<<<gq, 256, 0, stream>>>(xb, wVb, bv + (long long)i * HH * DKK,
        nullptr, vb, MM, DKK, DD, DD, DKK, DKK,
        0LL, (long long)DD * DKK, (long long)MM * DKK, (long long)DKK, 1.f, 0, 0);

    // scores S[z] = Q[z] x K[z]^T / sqrt(DK)
    dim3 gs(LL / TN, LL / TM, HH * BB);
    k_gemm_bf16<<<gs, 256, 0, stream>>>(qb, kb, nullptr, Smat, nullptr,
        LL, LL, DKK, DKK, DKK, LL,
        (long long)LL * DKK, (long long)LL * DKK, (long long)LL * LL, 0LL,
        iscale, 0, /*transB=*/1);

    dim3 gsm(LL, HH * BB);
    k_softmax_mask<<<gsm, 256, 0, stream>>>(Smat, attb, pm);

    // O[z] = att[z] x V[z]
    dim3 gav(DKK / TN, LL / TM, HH * BB);
    k_gemm_bf16<<<gav, 256, 0, stream>>>(attb, vb, nullptr, nullptr, Ob,
        LL, DKK, LL, LL, DKK, DKK,
        (long long)LL * LL, (long long)LL * DKK, (long long)LL * DKK, 0LL,
        1.f, 0, 0);

    const long long ntot = (long long)HH * BB * LL * DKK;
    k_concat_heads<<<(unsigned)((ntot + 255) / 256), 256, 0, stream>>>(Ob, zc);

    // output projection + residual + LN1
    dim3 gop(DD / TN, MM / TM, 1);
    k_gemm_bf16<<<gop, 256, 0, stream>>>(zc, wOb, bo + (long long)i * DD,
        tf, nullptr, MM, DD, HH * DKK, HH * DKK, DD, DD,
        0LL, 0LL, 0LL, 0LL, 1.f, 0, 0);
    k_add_ln<<<MM, 256, 0, stream>>>(tf, xf, ln1g + (long long)i * DD,
                                     ln1b + (long long)i * DD, xf, xb);

    // FFN: relu(x W1 + b1) W2 + b2, residual + LN3
    dim3 gf1(DFFN / TN, MM / TM, 1);
    k_gemm_bf16<<<gf1, 256, 0, stream>>>(xb, w1b, b1 + (long long)i * DFFN,
        nullptr, h1b, MM, DFFN, DD, DD, DFFN, DFFN,
        0LL, 0LL, 0LL, 0LL, 1.f, /*relu=*/1, 0);
    dim3 gf2(DD / TN, MM / TM, 1);
    k_gemm_bf16<<<gf2, 256, 0, stream>>>(h1b, w2b, b2 + (long long)i * DD,
        tf, nullptr, MM, DD, DFFN, DFFN, DD, DD,
        0LL, 0LL, 0LL, 0LL, 1.f, 0, 0);
    k_add_ln<<<MM, 256, 0, stream>>>(tf, xf, ln3g + (long long)i * DD,
                                     ln3b + (long long)i * DD, xf, xb);
  }

  // logits = x @ Wout + bout -> d_out (fp32), 250 x 16 tiles
  dim3 gl(VV / TN, MM / TM, 1);
  k_gemm_bf16<<<gl, 256, 0, stream>>>(xb, woutb, bout, out, nullptr,
      MM, VV, DD, DD, VV, VV, 0LL, 0LL, 0LL, 0LL, 1.f, 0, 0);
}
